// SpikingWorldModel_68401649156480
// MI455X (gfx1250) — matrix operand order repaired and checked by reading, not compiled, not run on hardware
//
#include <hip/hip_runtime.h>

typedef __attribute__((ext_vector_type(16))) _Float16 v16h;
typedef __attribute__((ext_vector_type(8)))  _Float16 v8h;
typedef __attribute__((ext_vector_type(8)))  float    v8f;

namespace {
constexpr int   kD       = 512;
constexpr int   kFF      = 1024;
constexpr int   kB       = 1024;
constexpr int   kSteps   = 64;
constexpr int   kA       = 32;
constexpr int   kL       = 4;
constexpr float kSfTh    = 2.0f;
constexpr int   kLdh     = 520;   // hs row stride (halves), 16B-multiple
constexpr int   kLds     = 1040;  // ss row stride (halves), 16B-multiple
constexpr int   kThreads = 512;   // 16 waves
constexpr int   kWaves   = 16;
constexpr int   kCT      = 2;     // 16-col tiles per wave over D   (16*2*16=512)
constexpr int   kFT      = 4;     // 16-col tiles per wave over FF  (16*4*16=1024)
}

// A-fragment loader: 16x32 f16 tile from LDS (row stride LDA halves).
// Contiguous k order (same as the B loads, not the ISA table order): lane m=lane&15;
// K = {16*half + 0..7} and {16*half + 8..15} -> two 16B LDS loads.
template <int LDA>
__device__ __forceinline__ v16h load_afrag(const _Float16* As, int kk) {
  const int lane = threadIdx.x & 31;
  const int half = lane >> 4;
  const int nlo  = lane & 15;
  union { v16h v; v8h h[2]; } ua;
  ua.h[0] = *(const v8h*)(As + nlo * LDA + kk + 16 * half);
  ua.h[1] = *(const v8h*)(As + nlo * LDA + kk + 16 * half + 8);
  return ua.v;
}

// ---------------------------------------------------------------------------
// One 16x16 f32 tile = A(16xKDIM f16, LDS) * B^T(NxKDIM f16, global n-major).
// B 32x16 f16 fragment: lane n=lane&15; K block 16*half.. -> one contiguous
// 32B load from the n-major weight row.
// ---------------------------------------------------------------------------
template <int KDIM, int LDA>
__device__ __forceinline__ v8f wmma_rowtile(const _Float16* As,
                                            const _Float16* __restrict__ BT,
                                            int n0) {
  const int lane = threadIdx.x & 31;
  const int half = lane >> 4;
  const int nlo  = lane & 15;
  v8f acc = {};
  const _Float16* brow = BT + (size_t)(n0 + nlo) * KDIM;
  for (int kk = 0; kk < KDIM; kk += 32) {
    v16h af = load_afrag<LDA>(As, kk);
    v16h bf = *(const v16h*)(brow + kk + 16 * half);
    if (kk + 32 < KDIM)
      __builtin_prefetch(brow + kk + 32 + 16 * half, 0, 3);
    acc = __builtin_amdgcn_wmma_f32_16x16x32_f16(
        false, af, false, bf, (short)0, acc, false, false);
  }
  return acc;
}

// LayerNorm of the register-resident residual (wave-sliced C-fragment layout)
// into LDS f16. Deterministic: shuffle partials -> fixed-order LDS tree sum.
__device__ __forceinline__ void ln_to_lds(const v8f xreg[kCT], _Float16* dsth,
                                          const float* __restrict__ g,
                                          const float* __restrict__ b,
                                          float* rp /*[kWaves][16][2]*/,
                                          float* st /*[16][2]*/) {
  const int wave = threadIdx.x >> 5;
  const int lane = threadIdx.x & 31;
  const int half = lane >> 4;
  const int nlo  = lane & 15;
  float s[8], q[8];
#pragma unroll
  for (int r = 0; r < 8; ++r) {
    float a = 0.f, qq = 0.f;
#pragma unroll
    for (int c = 0; c < kCT; ++c) { float v = xreg[c][r]; a += v; qq += v * v; }
#pragma unroll
    for (int m = 1; m < 16; m <<= 1) {
      a  += __shfl_xor(a, m, 32);
      qq += __shfl_xor(qq, m, 32);
    }
    s[r] = a; q[r] = qq;
  }
  if (nlo == 0) {
#pragma unroll
    for (int r = 0; r < 8; ++r) {
      const int row = r + 8 * half;
      rp[(wave * 16 + row) * 2 + 0] = s[r];
      rp[(wave * 16 + row) * 2 + 1] = q[r];
    }
  }
  __syncthreads();
  if (threadIdx.x < 16) {
    float ts = 0.f, tq = 0.f;
#pragma unroll
    for (int w = 0; w < kWaves; ++w) {
      ts += rp[(w * 16 + threadIdx.x) * 2 + 0];
      tq += rp[(w * 16 + threadIdx.x) * 2 + 1];
    }
    const float mean = ts * (1.0f / kD);
    const float var  = tq * (1.0f / kD) - mean * mean;
    st[threadIdx.x * 2 + 0] = mean;
    st[threadIdx.x * 2 + 1] = rsqrtf(var + 1e-5f);
  }
  __syncthreads();
#pragma unroll
  for (int r = 0; r < 8; ++r) {
    const int row = r + 8 * half;
    const float mean = st[row * 2 + 0];
    const float rstd = st[row * 2 + 1];
#pragma unroll
    for (int c = 0; c < kCT; ++c) {
      const int col = wave * (16 * kCT) + c * 16 + nlo;
      dsth[row * kLdh + col] =
          (_Float16)((xreg[c][r] - mean) * rstd * g[col] + b[col]);
    }
  }
  __syncthreads();
}

// ---------------------------------------------------------------------------
// Whole 64-step scan in ONE kernel. Block = 16 batch rows (512 threads,
// 16 waves); wave = 32 columns of the residual stream, held in registers as
// WMMA C-fragments for the entire recurrence (state never touches memory).
// ---------------------------------------------------------------------------
__global__ __launch_bounds__(kThreads) void scan_kernel(
    const float* __restrict__ init_state, const float* __restrict__ act,
    const _Float16* __restrict__ WaT, const float* __restrict__ ba,
    const float* __restrict__ ln1g, const float* __restrict__ ln1b,
    const _Float16* __restrict__ WvT, const float* __restrict__ bv,
    const _Float16* __restrict__ WoT, const float* __restrict__ bo,
    const float* __restrict__ ln2g, const float* __restrict__ ln2b,
    const _Float16* __restrict__ Wf1T, const float* __restrict__ bf1,
    const _Float16* __restrict__ Wf2T, const float* __restrict__ bf2,
    const float* __restrict__ lifth, const float* __restrict__ Wr,
    const float* __restrict__ brs, float* __restrict__ states,
    float* __restrict__ rewards, float* __restrict__ finals) {
  __shared__ __attribute__((aligned(16))) _Float16 hs[16 * kLdh];
  __shared__ __attribute__((aligned(16))) _Float16 ss[16 * kLds];
  __shared__ float rp[kWaves * 16 * 2];
  __shared__ float st[16 * 2];

  const int row0 = blockIdx.x * 16;
  const int wave = threadIdx.x >> 5;
  const int lane = threadIdx.x & 31;
  const int half = lane >> 4;
  const int nlo  = lane & 15;
  const int col0 = wave * (16 * kCT);   // wave's 32 residual columns

  // This lane's fixed residual columns and per-column constants.
  int   cols[kCT];
  float bav[kCT], thv[kCT], wrv[kCT];
#pragma unroll
  for (int c = 0; c < kCT; ++c) {
    cols[c] = col0 + c * 16 + nlo;
    bav[c]  = ba[cols[c]];
    thv[c]  = lifth[cols[c]];
    wrv[c]  = Wr[cols[c]];
  }
  const float brv = brs[0];

  // Step-invariant embed B-fragments (K = kA = 32 -> exactly one fragment
  // per tile). Preloaded ONCE; kept live across the whole scan so the embed
  // WMMA needs no per-step weight traffic.
  v16h wafrag[kCT];
#pragma unroll
  for (int c = 0; c < kCT; ++c)
    wafrag[c] = *(const v16h*)(WaT + (size_t)cols[c] * kA + 16 * half);

  // Residual stream lives here for all 64 steps.
  v8f xreg[kCT];
#pragma unroll
  for (int c = 0; c < kCT; ++c)
#pragma unroll
    for (int r = 0; r < 8; ++r)
      xreg[c][r] = init_state[(size_t)(row0 + r + 8 * half) * kD + cols[c]];

  for (int step = 0; step < kSteps; ++step) {
    // ---- embed: x += a @ Wa + ba (single WMMA per tile).
    // Action tile 16x32 = exactly one element per thread: one coalesced load.
    {
      const int row = threadIdx.x >> 5, i = threadIdx.x & 31;
      hs[row * 32 + i] =
          (_Float16)act[((size_t)(row0 + row) * kSteps + step) * kA + i];
    }
    __syncthreads();
    {
      const v16h af = load_afrag<kA>(hs, 0);
#pragma unroll
      for (int c = 0; c < kCT; ++c) {
        v8f acc = {};
        acc = __builtin_amdgcn_wmma_f32_16x16x32_f16(
            false, af, false, wafrag[c], (short)0, acc, false, false);
#pragma unroll
        for (int r = 0; r < 8; ++r) xreg[c][r] += acc[r] + bav[c];
      }
    }
    __syncthreads();

    for (int l = 0; l < kL; ++l) {
      const _Float16* wv = WvT  + (size_t)l * kD * kD;
      const _Float16* wo = WoT  + (size_t)l * kD * kD;
      const _Float16* w1 = Wf1T + (size_t)l * kD * kFF;
      const _Float16* w2 = Wf2T + (size_t)l * kFF * kD;

      // ---- attention block (softmax over length-1 axis == identity):
      //      x += (LN1(x) @ Wv + bv) @ Wo + bo
      ln_to_lds(xreg, hs, ln1g + l * kD, ln1b + l * kD, rp, st);
#pragma unroll
      for (int c = 0; c < kCT; ++c) {
        const int n0 = col0 + c * 16;
        v8f acc = wmma_rowtile<kD, kLdh>(hs, wv, n0);
        const float bias = bv[l * kD + n0 + nlo];
#pragma unroll
        for (int r = 0; r < 8; ++r)
          ss[(r + 8 * half) * kLdh + n0 + nlo] = (_Float16)(acc[r] + bias);
      }
      __syncthreads();
#pragma unroll
      for (int c = 0; c < kCT; ++c) {
        const int n0 = col0 + c * 16;
        v8f acc = wmma_rowtile<kD, kLdh>(ss, wo, n0);
        const float bias = bo[l * kD + n0 + nlo];
#pragma unroll
        for (int r = 0; r < 8; ++r) xreg[c][r] += acc[r] + bias;
      }
      __syncthreads();

      // ---- FFN block: x += spike(LN2(x) @ Wf1 + bf1 - 2) @ Wf2 + bf2
      ln_to_lds(xreg, hs, ln2g + l * kD, ln2b + l * kD, rp, st);
#pragma unroll
      for (int c = 0; c < kFT; ++c) {
        const int n0 = wave * (16 * kFT) + c * 16;
        v8f acc = wmma_rowtile<kD, kLdh>(hs, w1, n0);
        const float bias = bf1[l * kFF + n0 + nlo] - kSfTh;
#pragma unroll
        for (int r = 0; r < 8; ++r) {
          const float v = acc[r] + bias;
          ss[(r + 8 * half) * kLds + n0 + nlo] =
              (v > 0.0f) ? (_Float16)1.0f : (_Float16)0.0f;
        }
      }
      __syncthreads();
#pragma unroll
      for (int c = 0; c < kCT; ++c) {
        const int n0 = col0 + c * 16;
        v8f acc = wmma_rowtile<kFF, kLds>(ss, w2, n0);
        const float bias = bf2[l * kD + n0 + nlo];
#pragma unroll
        for (int r = 0; r < 8; ++r) xreg[c][r] += acc[r] + bias;
      }
      __syncthreads();
    }

    // ---- LIF spike, outputs, reward (state stays in xreg for next step) ----
    float pr[8];
#pragma unroll
    for (int r = 0; r < 8; ++r) pr[r] = 0.f;
#pragma unroll
    for (int c = 0; c < kCT; ++c) {
#pragma unroll
      for (int r = 0; r < 8; ++r) {
        const int row = r + 8 * half;
        const float sv = ((xreg[c][r] - thv[c]) > 0.0f) ? 1.0f : 0.0f;
        xreg[c][r] = sv;
        states[((size_t)(row0 + row) * kSteps + step) * kD + cols[c]] = sv;
        if (step == kSteps - 1)
          finals[(size_t)(row0 + row) * kD + cols[c]] = sv;
        pr[r] += sv * wrv[c];
      }
    }
#pragma unroll
    for (int r = 0; r < 8; ++r) {
#pragma unroll
      for (int m = 1; m < 16; m <<= 1) pr[r] += __shfl_xor(pr[r], m, 32);
    }
    if (nlo == 0) {
#pragma unroll
      for (int r = 0; r < 8; ++r) rp[(wave * 16 + r + 8 * half) * 2] = pr[r];
    }
    __syncthreads();
    if (threadIdx.x < 16) {
      float t = 0.f;
#pragma unroll
      for (int w = 0; w < kWaves; ++w) t += rp[(w * 16 + threadIdx.x) * 2];
      rewards[(size_t)(row0 + threadIdx.x) * kSteps + step] = t + brv;
    }
    __syncthreads();
  }
}

// dst[n*K + k] = (f16) src[k*N + n]  — weight convert+transpose to n-major.
__global__ __launch_bounds__(256) void convt_kernel(
    const float* __restrict__ src, _Float16* __restrict__ dst, int K, int N) {
  const int idx = blockIdx.x * 256 + threadIdx.x;
  if (idx >= K * N) return;
  const int n = idx / K;
  const int k = idx - n * K;
  dst[idx] = (_Float16)src[k * N + n];
}

extern "C" void kernel_launch(void* const* d_in, const int* in_sizes, int n_in,
                              void* d_out, int out_size, void* d_ws,
                              size_t ws_size, hipStream_t stream) {
  (void)in_sizes; (void)n_in; (void)out_size; (void)ws_size;

  const float* initial_state = (const float*)d_in[0];
  const float* act   = (const float*)d_in[1];
  const float* Wa    = (const float*)d_in[2];
  const float* ba    = (const float*)d_in[3];
  const float* ln1g  = (const float*)d_in[4];
  const float* ln1b  = (const float*)d_in[5];
  // d_in[6..9] = Wq,bq,Wk,bk: dead (softmax over a length-1 axis == 1).
  const float* Wv    = (const float*)d_in[10];
  const float* bv    = (const float*)d_in[11];
  const float* Wo    = (const float*)d_in[12];
  const float* bo    = (const float*)d_in[13];
  const float* ln2g  = (const float*)d_in[14];
  const float* ln2b  = (const float*)d_in[15];
  const float* Wf1   = (const float*)d_in[16];
  const float* bf1   = (const float*)d_in[17];
  const float* Wf2   = (const float*)d_in[18];
  const float* bf2   = (const float*)d_in[19];
  const float* lifth = (const float*)d_in[20];
  const float* Wr    = (const float*)d_in[21];
  const float* br    = (const float*)d_in[22];

  size_t off = 0;
  auto carve = [&](size_t bytes) {
    void* p = (char*)d_ws + off;
    off += (bytes + 255) & ~(size_t)255;
    return p;
  };
  _Float16* WaT  = (_Float16*)carve((size_t)kA * kD * 2);
  _Float16* WvT  = (_Float16*)carve((size_t)kL * kD * kD * 2);
  _Float16* WoT  = (_Float16*)carve((size_t)kL * kD * kD * 2);
  _Float16* Wf1T = (_Float16*)carve((size_t)kL * kD * kFF * 2);
  _Float16* Wf2T = (_Float16*)carve((size_t)kL * kFF * kD * 2);

  float* states  = (float*)d_out;                      // B*STEPS*D
  float* rewards = states + (size_t)kB * kSteps * kD;  // B*STEPS
  float* finals  = rewards + (size_t)kB * kSteps;      // B*D

  // One-time weight convert+transpose to f16 n-major (L2-resident after).
  const int gDD = (kD * kD + 255) / 256;
  const int gDF = (kD * kFF + 255) / 256;
  convt_kernel<<<(kA * kD + 255) / 256, 256, 0, stream>>>(Wa, WaT, kA, kD);
  for (int l = 0; l < kL; ++l) {
    convt_kernel<<<gDD, 256, 0, stream>>>(Wv + (size_t)l * kD * kD,
                                          WvT + (size_t)l * kD * kD, kD, kD);
    convt_kernel<<<gDD, 256, 0, stream>>>(Wo + (size_t)l * kD * kD,
                                          WoT + (size_t)l * kD * kD, kD, kD);
    convt_kernel<<<gDF, 256, 0, stream>>>(Wf1 + (size_t)l * kD * kFF,
                                          Wf1T + (size_t)l * kD * kFF, kD, kFF);
    convt_kernel<<<gDF, 256, 0, stream>>>(Wf2 + (size_t)l * kFF * kD,
                                          Wf2T + (size_t)l * kFF * kD, kFF, kD);
  }

  // Entire 64-step recurrence: one kernel, 64 blocks x 512 threads.
  scan_kernel<<<kB / 16, kThreads, 0, stream>>>(
      initial_state, act, WaT, ba, ln1g, ln1b, WvT, bv, WoT, bo, ln2g, ln2b,
      Wf1T, bf1, Wf2T, bf2, lifth, Wr, br, states, rewards, finals);
}
